// CausalSCMLayer_88261577933267
// MI455X (gfx1250) — matrix operand order repaired and checked by reading, not compiled, not run on hardware
//
#include <hip/hip_runtime.h>
#include <hip/hip_bf16.h>

// z_causal = Z * (I - A)^{-1},  A = tril(A_raw, -1), N = 256, BATCH = 131072.
// Step 1: invert unit-lower-triangular (I - A) once (tiny).
// Step 2: big GEMM with bf16x3 split-precision WMMA (fp32 accumulate).

#define NVARS 256

typedef __attribute__((ext_vector_type(16))) __bf16 v16bf;
typedef __attribute__((ext_vector_type(8)))  float  v8f;

// Native RNE fp32 -> bf16 split: f ~= hi + lo (each bf16).
__device__ __forceinline__ void split_bf16(float f, __bf16& h, __bf16& l) {
    h = (__bf16)f;
    l = (__bf16)(f - (float)h);
}

// Interleaved W^T layout (ushort index):  row n (0..255) has 16 chunks of
// 16 K-values; chunk c holds hi[16] then lo[16] adjacent:
//   hi(n, c, k%16) at (n*16 + c)*32 + (k%16)
//   lo(n, c, k%16) at (n*16 + c)*32 + 16 + (k%16)
// => a B-fragment's bhi/blo are 32B runs sharing one base (DS imm offsets).
#define WT_POS(n, c) (((n) * 16 + (c)) * 32)

// ---------------------------------------------------------------------------
// Kernel A: W = (I - A)^{-1} by per-column forward substitution.
// One block, 256 threads; thread j owns column j (no inter-thread deps).
// W[i][j] = delta_ij + sum_{k<i} A[i,k] * W[k][j]   (A strictly lower)
// Emits the interleaved bf16 hi/lo W^T buffer (256 KB) for the GEMM.
// ---------------------------------------------------------------------------
__global__ void scm_invert(const float* __restrict__ A_raw,
                           unsigned short* __restrict__ wt) {
    extern __shared__ float W[];            // 256*256 fp32 = 256 KB LDS
    const int j = threadIdx.x;              // column owned by this thread
    for (int i = 0; i < NVARS; ++i) {
        float s = (i == j) ? 1.0f : 0.0f;
        const float* arow = A_raw + (size_t)i * NVARS;   // uniform address
        for (int k = 0; k < i; ++k) {
            s += arow[k] * W[k * NVARS + j];             // bank-conflict free
        }
        W[i * NVARS + j] = s;
    }
    // write interleaved bf16 hi/lo W^T (row n = j)
    for (int k = 0; k < NVARS; ++k) {
        float w = W[k * NVARS + j];
        __bf16 h, l;
        split_bf16(w, h, l);
        const int c = k >> 4;
        const int r = k & 15;
        wt[WT_POS(j, c) + r]      = __builtin_bit_cast(unsigned short, h);
        wt[WT_POS(j, c) + 16 + r] = __builtin_bit_cast(unsigned short, l);
    }
}

// ---------------------------------------------------------------------------
// Kernel B: OUT = Z * W, split-precision bf16 WMMA (hi*hi + hi*lo + lo*hi),
// fp32 accumulation. Block = 256 threads (8 waves), covers 64 rows x 256 cols.
// Wave w: M-strip (w&3)*16, N-half (w>>2)*128 -> 8 accumulator tiles (64 VGPR).
// Interleaved W^T (256 KB) staged in LDS once per block; Z is read from HBM
// exactly once (memory-roofline optimal: 128 MB in + 128 MB out ~ 11 us).
// ---------------------------------------------------------------------------
__global__ void scm_gemm(const float* __restrict__ Z,
                         const unsigned short* __restrict__ wt,
                         float* __restrict__ out) {
    extern __shared__ unsigned short lds[];   // interleaved hi/lo W^T, 256 KB
    const int tid = threadIdx.x;

    // Cooperative stage of the interleaved W^T buffer (256 KB) into LDS.
    {
        const uint4* src = (const uint4*)wt;
        uint4* dst = (uint4*)lds;
        #pragma unroll 4
        for (int i = tid; i < 16384; i += 256) {   // 16384 uint4 = 256 KB
            dst[i] = src[i];
        }
    }
    __syncthreads();

    const int wave   = tid >> 5;
    const int lane   = tid & 31;
    const int l16    = lane & 15;
    const int khalf  = lane >> 4;          // 0: K 0-15 half, 1: K 16-31 half
    const int mstrip = wave & 3;
    const int nhalf  = wave >> 2;
    const int n0     = nhalf * 128;

    const long rowA  = (long)blockIdx.x * 64 + mstrip * 16 + l16;  // A-frag row
    const float* zrow = Z + rowA * NVARS;

    v8f acc[8] = {};                        // 8 N-tiles x 16x16 fp32

    for (int k0 = 0; k0 < NVARS; k0 += 32) {
        if (k0 + 32 < NVARS)
            __builtin_prefetch(zrow + k0 + 32 + khalf * 8, 0, 0);  // global_prefetch_b8

        // --- A fragment (16x32 bf16): lane holds row M=l16; contiguous K order
        // elements 0..15 = K {k0 + khalf*16 + 0..15} (matches B)
        const float* p0 = zrow + k0 + khalf * 16;
        const float* p1 = p0 + 8;
        float4 f0 = *(const float4*)(p0);
        float4 f1 = *(const float4*)(p0 + 4);
        float4 f2 = *(const float4*)(p1);
        float4 f3 = *(const float4*)(p1 + 4);
        float v[16] = { f0.x, f0.y, f0.z, f0.w, f1.x, f1.y, f1.z, f1.w,
                        f2.x, f2.y, f2.z, f2.w, f3.x, f3.y, f3.z, f3.w };
        v16bf ahi, alo;
        #pragma unroll
        for (int i = 0; i < 16; ++i) {
            __bf16 h, l;
            split_bf16(v[i], h, l);
            ahi[i] = h;
            alo[i] = l;
        }

        // --- 8 N-tiles: B fragment (32x16 bf16): lane = column n0+t*16+l16,
        // lanes 0-15 hold K k0..k0+15, lanes 16-31 hold K k0+16..k0+31.
        // bhi/blo are adjacent 32B runs -> 4 ds_load_b128 from one base.
        const int c = (k0 >> 4) + khalf;
        #pragma unroll
        for (int t = 0; t < 8; ++t) {
            const int n = n0 + t * 16 + l16;
            const unsigned short* bp = lds + WT_POS(n, c);
            v16bf bhi = *(const v16bf*)bp;
            v16bf blo = *(const v16bf*)(bp + 16);
            acc[t] = __builtin_amdgcn_wmma_f32_16x16x32_bf16(
                         false, ahi, false, bhi, (short)0, acc[t], false, false);
            acc[t] = __builtin_amdgcn_wmma_f32_16x16x32_bf16(
                         false, alo, false, bhi, (short)0, acc[t], false, false);
            acc[t] = __builtin_amdgcn_wmma_f32_16x16x32_bf16(
                         false, ahi, false, blo, (short)0, acc[t], false, false);
        }
    }

    // --- Store: C/D 16x16 fp32 layout: VGPR r -> M = r (lanes 0-15), r+8 (lanes 16-31)
    const long rbase = (long)blockIdx.x * 64 + mstrip * 16 + khalf * 8;
    #pragma unroll
    for (int t = 0; t < 8; ++t) {
        const int n = n0 + t * 16 + l16;
        #pragma unroll
        for (int r = 0; r < 8; ++r) {
            out[(rbase + r) * NVARS + n] = acc[t][r];
        }
    }
}

extern "C" void kernel_launch(void* const* d_in, const int* in_sizes, int n_in,
                              void* d_out, int out_size, void* d_ws, size_t ws_size,
                              hipStream_t stream) {
    const float* z_exo  = (const float*)d_in[0];   // [BATCH, 256] fp32
    const float* A_raw  = (const float*)d_in[1];   // [256, 256]  fp32
    float*       outp   = (float*)d_out;           // [BATCH, 256] fp32

    // workspace: interleaved bf16 hi/lo W^T, 2*256*256 ushort = 256 KB
    unsigned short* wt = (unsigned short*)d_ws;

    const int rows   = in_sizes[0] / NVARS;        // 131072
    const int blocks = rows / 64;                  // 2048

    scm_invert<<<1, 256, NVARS * NVARS * sizeof(float), stream>>>(A_raw, wt);
    scm_gemm<<<blocks, 256, 2 * NVARS * NVARS * sizeof(unsigned short), stream>>>(
        z_exo, wt, outp);
}
